// EdgeSetUpdate_56186762167005
// MI455X (gfx1250) — matrix operand order repaired and checked by reading, not compiled, not run on hardware
//
#include <hip/hip_runtime.h>

// ---------------------------------------------------------------------------
// EdgeSetUpdate: out = relu(concat(edge_feat, node[src], node[tgt]) @ W + b)
//   E=300000, D=256, K=768.  Compute-bound in f32 (~380us) vs ~28us HBM ->
//   3xBF16 hi/lo-split WMMA (~140us compute, near-f32 accuracy).
//   node_feat (51MB) + W (0.75MB) are kept L2-resident by streaming
//   edge_feat/out with NT cache hints (L2 = 192MB).
//   Block tile: 128 edges x 256 cols; wave tile: 32 x 128 (16 acc tiles) so
//   each B-fragment load feeds 6 WMMAs (halves B bandwidth per FLOP).
// ---------------------------------------------------------------------------

typedef __bf16 bf16_t;
typedef __attribute__((ext_vector_type(16))) __bf16 v16bf;
typedef __attribute__((ext_vector_type(8)))  __bf16 v8bf;
typedef __attribute__((ext_vector_type(8)))  float   v8f;
typedef __attribute__((ext_vector_type(4)))  float   v4f;

#define DD    256          // feature width
#define KTOT  768          // 3*D
#define MT    128          // edge rows per block
#define LDA   40           // LDS row stride in bf16 (80B rows, 16B aligned)

// ---------------------------------------------------------------------------
// Kernel 1: split W (f32 [768,256] row-major) into bf16 hi/lo, pre-swizzled
// into WMMA B-fragment layout: fragment (kt,nt), lane l=(hl,c) holds
// W[kt*32 + 16*hl + j][nt*16 + c] for j=0..15 contiguously.
// ---------------------------------------------------------------------------
__global__ __launch_bounds__(256)
void prep_w_kernel(const float* __restrict__ W,
                   bf16_t* __restrict__ whi,
                   bf16_t* __restrict__ wlo)
{
    int id   = blockIdx.x * 256 + threadIdx.x;   // 0 .. 768*256-1
    int j    = id & 15;
    int lane = (id >> 4) & 31;
    int frag = id >> 9;                          // kt*16 + nt
    int nt   = frag & 15;
    int kt   = frag >> 4;
    int k = kt * 32 + (lane >> 4) * 16 + j;
    int n = nt * 16 + (lane & 15);
    float w = W[k * DD + n];
    bf16_t h = (bf16_t)w;
    whi[id] = h;
    wlo[id] = (bf16_t)(w - (float)h);
}

// ---------------------------------------------------------------------------
// Kernel 2: fused gather + GEMM + bias + relu.
// 8 waves: wave_m = wid&3 (32-row strip), wave_n = wid>>2 (128-col half).
// ---------------------------------------------------------------------------
__global__ __launch_bounds__(256)
void edge_gemm_kernel(const float* __restrict__ edge_feat,
                      const float* __restrict__ node_feat,
                      const int*   __restrict__ src_idx,
                      const int*   __restrict__ tgt_idx,
                      const bf16_t* __restrict__ whi,
                      const bf16_t* __restrict__ wlo,
                      const float* __restrict__ bias,
                      float* __restrict__ out,
                      int nEdges)
{
    __shared__ __align__(16) bf16_t a_hi[MT * LDA];
    __shared__ __align__(16) bf16_t a_lo[MT * LDA];

    const int t      = threadIdx.x;
    const int lane   = t & 31;
    const int wid    = t >> 5;       // 0..7
    const int wave_m = wid & 3;      // 32-row strip
    const int wave_n = wid >> 2;     // 128-col half
    const int hl     = lane >> 4;
    const int lr     = lane & 15;

    const int m0 = blockIdx.x * MT;

    // Loader role: thread t stages 16 contiguous floats of row (t&127),
    // half (t>>7) of the 32-float k-chunk.
    const int lrow = t & 127;
    const int lseg = t >> 7;         // 0..1
    int er = m0 + lrow;
    if (er >= nEdges) er = nEdges - 1;   // clamp; stores are guarded
    const float* p0 = edge_feat + (size_t)er * DD;
    const float* p1 = node_feat + (size_t)src_idx[er] * DD;
    const float* p2 = node_feat + (size_t)tgt_idx[er] * DD;

    v8f acc[16];
    #pragma unroll
    for (int i = 0; i < 16; ++i)
        #pragma unroll
        for (int v = 0; v < 8; ++v) acc[i][v] = 0.0f;

    // One k-chunk (K=32): stage A tile hi/lo to LDS, read fragments, 48 WMMAs.
    auto compute_chunk = [&](v4f x0, v4f x1, v4f x2, v4f x3, int kc) {
        __syncthreads();   // previous chunk's fragment reads complete
        const int abase = lrow * LDA + lseg * 16;
        #pragma unroll
        for (int j = 0; j < 16; ++j) {
            float xv = (j < 4) ? x0[j] : (j < 8) ? x1[j - 4]
                     : (j < 12) ? x2[j - 8] : x3[j - 12];
            bf16_t h = (bf16_t)xv;
            a_hi[abase + j] = h;
            a_lo[abase + j] = (bf16_t)(xv - (float)h);
        }
        __syncthreads();

        // A fragments (contiguous k order, not the ISA table order): lane holds K in
        // [16*hl,16*hl+16) contiguously (matches B pre-swizzle).
        v16bf afh[2], afl[2];
        #pragma unroll
        for (int ms = 0; ms < 2; ++ms) {
            const int ro = (wave_m * 32 + ms * 16 + lr) * LDA + 16 * hl;
            v8bf c0 = *(const v8bf*)&a_hi[ro];
            v8bf c1 = *(const v8bf*)&a_hi[ro + 8];
            v8bf d0 = *(const v8bf*)&a_lo[ro];
            v8bf d1 = *(const v8bf*)&a_lo[ro + 8];
            #pragma unroll
            for (int j = 0; j < 8; ++j) {
                afh[ms][j] = c0[j]; afh[ms][8 + j] = c1[j];
                afl[ms][j] = d0[j]; afl[ms][8 + j] = d1[j];
            }
        }

        #pragma unroll
        for (int i = 0; i < 8; ++i) {
            const int nt = wave_n * 8 + i;
            const size_t fo = (size_t)(kc * 16 + nt) * 512 + (size_t)lane * 16;
            v16bf bh = *(const v16bf*)(whi + fo);
            v16bf bl = *(const v16bf*)(wlo + fo);
            #pragma unroll
            for (int ms = 0; ms < 2; ++ms) {
                v8f a = acc[ms * 8 + i];
                // 3xBF16 split: hi*hi + lo*hi + hi*lo
                a = __builtin_amdgcn_wmma_f32_16x16x32_bf16(
                        false, afh[ms], false, bh, (short)0, a, false, false);
                a = __builtin_amdgcn_wmma_f32_16x16x32_bf16(
                        false, afl[ms], false, bh, (short)0, a, false, false);
                a = __builtin_amdgcn_wmma_f32_16x16x32_bf16(
                        false, afh[ms], false, bl, (short)0, a, false, false);
                acc[ms * 8 + i] = a;
            }
        }
    };

    // Region 1: edge_feat (streams once through HBM -> NT loads protect L2).
    #pragma unroll 1
    for (int c = 0; c < 8; ++c) {
        const v4f* p = (const v4f*)(p0 + c * 32 + lseg * 16);
        v4f x0 = __builtin_nontemporal_load(p);
        v4f x1 = __builtin_nontemporal_load(p + 1);
        v4f x2 = __builtin_nontemporal_load(p + 2);
        v4f x3 = __builtin_nontemporal_load(p + 3);
        compute_chunk(x0, x1, x2, x3, c);
    }
    // Region 2: node_feat[src] (L2-resident gather, default RT caching).
    #pragma unroll 1
    for (int c = 0; c < 8; ++c) {
        const v4f* p = (const v4f*)(p1 + c * 32 + lseg * 16);
        compute_chunk(p[0], p[1], p[2], p[3], 8 + c);
    }
    // Region 3: node_feat[tgt].
    #pragma unroll 1
    for (int c = 0; c < 8; ++c) {
        const v4f* p = (const v4f*)(p2 + c * 32 + lseg * 16);
        compute_chunk(p[0], p[1], p[2], p[3], 16 + c);
    }

    // Epilogue: C/D layout — VGPR v: lanes 0-15 -> M=v, lanes 16-31 -> M=8+v.
    // Output streams once -> NT stores.
    #pragma unroll
    for (int ms = 0; ms < 2; ++ms) {
        #pragma unroll
        for (int i = 0; i < 8; ++i) {
            const int col = wave_n * 128 + i * 16 + lr;
            const float bv = bias[col];
            const int rbase = m0 + wave_m * 32 + ms * 16 + hl * 8;
            #pragma unroll
            for (int v = 0; v < 8; ++v) {
                const int row = rbase + v;
                if (row < nEdges) {
                    float r = acc[ms * 8 + i][v] + bv;
                    r = r > 0.0f ? r : 0.0f;
                    __builtin_nontemporal_store(r, &out[(size_t)row * DD + col]);
                }
            }
        }
    }
}

// ---------------------------------------------------------------------------
extern "C" void kernel_launch(void* const* d_in, const int* in_sizes, int n_in,
                              void* d_out, int out_size, void* d_ws, size_t ws_size,
                              hipStream_t stream)
{
    const float* edge_feat = (const float*)d_in[0];
    const float* node_feat = (const float*)d_in[1];
    const int*   src_idx   = (const int*)d_in[2];
    const int*   tgt_idx   = (const int*)d_in[3];
    const float* W         = (const float*)d_in[4];
    const float* bias      = (const float*)d_in[5];
    float*       out       = (float*)d_out;
    const int nEdges = in_sizes[2];   // src_idx count == E

    // Workspace: pre-swizzled bf16 hi/lo W (2 * 768*256*2B = 786 KB).
    bf16_t* whi = (bf16_t*)d_ws;
    bf16_t* wlo = (bf16_t*)((char*)d_ws + (size_t)KTOT * DD * sizeof(bf16_t));

    prep_w_kernel<<<(KTOT * DD) / 256, 256, 0, stream>>>(W, whi, wlo);

    const int nblk = (nEdges + MT - 1) / MT;
    edge_gemm_kernel<<<nblk, 256, 0, stream>>>(edge_feat, node_feat,
                                               src_idx, tgt_idx,
                                               whi, wlo, bias, out, nEdges);
}
